// Net_33088428048995
// MI455X (gfx1250) — compile-verified, matrix operand order checked by reading, not run on hardware
//
#include <hip/hip_runtime.h>
#include <hip/hip_bf16.h>

// ---------------------------------------------------------------------------
// Types for the CDNA5 WMMA builtin (wave32): v_wmma_f32_16x16x32_bf16
// ---------------------------------------------------------------------------
typedef __attribute__((ext_vector_type(16))) __bf16        v16bf;
typedef __attribute__((ext_vector_type(8)))  float         v8f;
typedef __attribute__((ext_vector_type(8)))  unsigned int  v8u;

static __device__ __forceinline__ unsigned short f2bf(float f) {
  union { float f; unsigned int u; } v;
  v.f = f;
  // round-to-nearest-even fp32 -> bf16
  return (unsigned short)((v.u + 0x7FFFu + ((v.u >> 16) & 1u)) >> 16);
}

// ---------------------------------------------------------------------------
// Fused depthwise(5,pad2)+ReLU -> depthwise(5,pad2)+ReLU -> MaxPool(5, stride2)
// x: (B, T, C) fp32, out: (B, To, C) fp32 with To = (T-5)/2 + 1.
// One thread per channel column; sliding windows live entirely in registers,
// so the input tensor is read exactly once from HBM (coalesced across C).
// ---------------------------------------------------------------------------
__global__ void dwconv2_relu_pool(const float* __restrict__ x,
                                  const float* __restrict__ wa, const float* __restrict__ ba,
                                  const float* __restrict__ wb, const float* __restrict__ bb,
                                  float* __restrict__ out,
                                  int T, int C, int To) {
  const int c = blockIdx.x * blockDim.x + threadIdx.x;
  const int b = blockIdx.y;
  if (c >= C) return;

  const float wa0 = wa[c * 5 + 0], wa1 = wa[c * 5 + 1], wa2 = wa[c * 5 + 2],
              wa3 = wa[c * 5 + 3], wa4 = wa[c * 5 + 4];
  const float wb0 = wb[c * 5 + 0], wb1 = wb[c * 5 + 1], wb2 = wb[c * 5 + 2],
              wb3 = wb[c * 5 + 3], wb4 = wb[c * 5 + 4];
  const float bac = ba[c], bbc = bb[c];

  float xw0 = 0.f, xw1 = 0.f, xw2 = 0.f, xw3 = 0.f, xw4 = 0.f;  // x[t-4..t]
  float zw0 = 0.f, zw1 = 0.f, zw2 = 0.f, zw3 = 0.f, zw4 = 0.f;  // y1[t-6..t-2]
  float pw0 = 0.f, pw1 = 0.f, pw2 = 0.f, pw3 = 0.f, pw4 = 0.f;  // y2[s-4..s]

  const float* __restrict__ xb = x + (size_t)b * T * C + c;
  float* __restrict__ ob = out + (size_t)b * To * C + c;

  for (int t = 0; t < T + 4; ++t) {
    const float xin = (t < T) ? xb[(size_t)t * C] : 0.f;
    xw0 = xw1; xw1 = xw2; xw2 = xw3; xw3 = xw4; xw4 = xin;

    if (t >= 2) {
      float y1 = 0.f;
      if (t < T + 2) {  // y1 index t-2 in [0, T)
        y1 = fmaxf(xw0 * wa0 + xw1 * wa1 + xw2 * wa2 + xw3 * wa3 + xw4 * wa4 + bac, 0.f);
      }
      zw0 = zw1; zw1 = zw2; zw2 = zw3; zw3 = zw4; zw4 = y1;
    }

    if (t >= 4) {
      const int s = t - 4;  // y2 index in [0, T)
      const float y2 =
          fmaxf(zw0 * wb0 + zw1 * wb1 + zw2 * wb2 + zw3 * wb3 + zw4 * wb4 + bbc, 0.f);
      pw0 = pw1; pw1 = pw2; pw2 = pw3; pw3 = pw4; pw4 = y2;
      if (s >= 4 && ((s & 1) == 0)) {
        const int tp = (s - 4) >> 1;  // pool output covering y2[2tp..2tp+4]
        ob[(size_t)tp * C] =
            fmaxf(fmaxf(fmaxf(pw0, pw1), fmaxf(pw2, pw3)), pw4);
      }
    }
  }
}

// ---------------------------------------------------------------------------
// Generic GEMM: D = act(A @ W^T + bias), A (M,K) fp32, W (N,K) fp32 (torch
// Linear layout), D (M,N) fp32. bf16 WMMA with fp32 accumulation.
//
// Block = 256 threads = 8 waves tiled 4(M) x 2(N); block tile 64x32, K-step 32.
// A/W tiles are staged through LDS with fully coalesced, branch-free global
// loads (row indices clamped for address safety -- those rows are never
// stored; K-tail zero-filled via select, no EXEC branching). Each lane then
// reads its ISA-7.12.2 WMMA fragment as two 16B-aligned ds_load_b128 per
// matrix: within K-group g, lane (half,r) owns the 8 contiguous bf16 at
// row r, cols g*16 + half*8 .. +7.
// ---------------------------------------------------------------------------
__global__ __launch_bounds__(256)
void gemm_bias_act_wmma(const float* __restrict__ A, const float* __restrict__ W,
                        const float* __restrict__ bias, float* __restrict__ Cout,
                        int M, int N, int K, int do_relu) {
  __shared__ unsigned short smA[64 * 32];  // [m][k] bf16
  __shared__ unsigned short smB[32 * 32];  // [n][k] bf16

  const int tid  = threadIdx.x;
  const int lane = tid & 31;
  const int wave = tid >> 5;
  const int wm   = wave & 3;
  const int wn   = wave >> 2;
  const int blk_m = blockIdx.x * 64;
  const int blk_n = blockIdx.y * 32;

  const int half = lane >> 4;  // selects K-octet within each 16-wide K-group
  const int r    = lane & 15;

  // Staging coordinates: flattened tile idx = tid + i*256; col = idx&31 (K),
  // row = idx>>5. Consecutive tids -> consecutive K -> coalesced 128B/wave.
  const int scol = tid & 31;
  const int srow = tid >> 5;

  // Clamped global row bases (clamping keeps addresses in-bounds; clamped
  // rows produce garbage that lands only in never-stored output rows/cols).
  const float* aRow[8];
#pragma unroll
  for (int i = 0; i < 8; ++i) {
    int gm = blk_m + srow + i * 8;
    if (gm > M - 1) gm = M - 1;
    aRow[i] = A + (size_t)gm * K;
  }
  const float* bRow[4];
#pragma unroll
  for (int i = 0; i < 4; ++i) {
    int gn = blk_n + srow + i * 8;
    if (gn > N - 1) gn = N - 1;
    bRow[i] = W + (size_t)gn * K;
  }

  v8f acc = {};
  for (int kb = 0; kb < K; kb += 32) {
    const int kc  = kb + scol;
    const int kcc = (kc < K) ? kc : (K - 1);   // clamp address
    const bool kok = (kc < K);

    __syncthreads();
    if (kb + 32 < K) {  // uniform cond: pull next K-slab (global_prefetch_b8)
      __builtin_prefetch(aRow[0] + kcc + 32, 0, 0);
      __builtin_prefetch(bRow[0] + kcc + 32, 0, 0);
    }
#pragma unroll
    for (int i = 0; i < 8; ++i) {
      float v = aRow[i][kcc];
      smA[(srow + i * 8) * 32 + scol] = f2bf(kok ? v : 0.f);
    }
#pragma unroll
    for (int i = 0; i < 4; ++i) {
      float v = bRow[i][kcc];
      smB[(srow + i * 8) * 32 + scol] = f2bf(kok ? v : 0.f);
    }
    __syncthreads();

    // Fragment fetch: 2x ds_load_b128 per matrix (16B aligned).
    const unsigned short* pa = &smA[(wm * 16 + r) * 32 + half * 8];
    const unsigned short* pb = &smB[(wn * 16 + r) * 32 + half * 8];
    const uint4 a0 = *(const uint4*)(pa);
    const uint4 a1 = *(const uint4*)(pa + 16);
    const uint4 b0 = *(const uint4*)(pb);
    const uint4 b1 = *(const uint4*)(pb + 16);
    const v8u au = {a0.x, a0.y, a0.z, a0.w, a1.x, a1.y, a1.z, a1.w};
    const v8u bu = {b0.x, b0.y, b0.z, b0.w, b1.x, b1.y, b1.z, b1.w};

    acc = __builtin_amdgcn_wmma_f32_16x16x32_bf16(
        false, __builtin_bit_cast(v16bf, au),
        false, __builtin_bit_cast(v16bf, bu),
        (short)0, acc, false, false);
  }

  // C/D layout: lane L holds N = tile_n + (L&15); VGPR j holds M = j + 8*(L>>4)
  const int cn = blk_n + wn * 16 + r;
  if (cn < N) {
    const float bv = bias ? bias[cn] : 0.f;
#pragma unroll
    for (int j = 0; j < 8; ++j) {
      const int cm = blk_m + wm * 16 + j + (half << 3);
      if (cm < M) {
        float o = acc[j] + bv;
        if (do_relu) o = fmaxf(o, 0.f);
        Cout[(size_t)cm * N + cn] = o;
      }
    }
  }
}

// ---------------------------------------------------------------------------
// z = concat(enc1 (B,430), e (B,100)) -> (B,530)
// ---------------------------------------------------------------------------
__global__ void concat_rows(const float* __restrict__ enc1, const float* __restrict__ e,
                            float* __restrict__ z, int B) {
  const int idx = blockIdx.x * blockDim.x + threadIdx.x;
  const int total = B * 530;
  if (idx >= total) return;
  const int b = idx / 530;
  const int j = idx - b * 530;
  z[idx] = (j < 430) ? enc1[b * 430 + j] : e[b * 100 + (j - 430)];
}

// ---------------------------------------------------------------------------
// Host-side launch
// ---------------------------------------------------------------------------
static void launch_gemm(const float* A, const float* W, const float* b, float* C,
                        int M, int N, int K, int relu, hipStream_t s) {
  dim3 grid((M + 63) / 64, (N + 31) / 32);
  gemm_bias_act_wmma<<<grid, 256, 0, s>>>(A, W, b, C, M, N, K, relu);
}

extern "C" void kernel_launch(void* const* d_in, const int* in_sizes, int n_in,
                              void* d_out, int out_size, void* d_ws, size_t ws_size,
                              hipStream_t stream) {
  (void)in_sizes; (void)n_in; (void)out_size; (void)ws_size;

  const float* x   = (const float*)d_in[0];
  const float* x2  = (const float*)d_in[1];
  const float* w1  = (const float*)d_in[2];  const float* b1 = (const float*)d_in[3];
  const float* w2  = (const float*)d_in[4];  const float* b2 = (const float*)d_in[5];
  const float* w3  = (const float*)d_in[6];  const float* b3 = (const float*)d_in[7];
  const float* w4  = (const float*)d_in[8];  const float* b4 = (const float*)d_in[9];
  const float* w5  = (const float*)d_in[10]; const float* b5 = (const float*)d_in[11];
  const float* w6  = (const float*)d_in[12]; const float* b6 = (const float*)d_in[13];
  const float* p1w = (const float*)d_in[14]; const float* p1b = (const float*)d_in[15];
  const float* p2w = (const float*)d_in[16]; const float* p2b = (const float*)d_in[17];
  const float* p3w = (const float*)d_in[18]; const float* p3b = (const float*)d_in[19];
  const float* l1w = (const float*)d_in[20]; const float* l1b = (const float*)d_in[21];
  const float* l2w = (const float*)d_in[22]; const float* l2b = (const float*)d_in[23];
  const float* l3w = (const float*)d_in[24]; const float* l3b = (const float*)d_in[25];
  const float* c1w = (const float*)d_in[26]; const float* c1b = (const float*)d_in[27];
  const float* c2w = (const float*)d_in[28]; const float* c2b = (const float*)d_in[29];
  const float* c3w = (const float*)d_in[30]; const float* c3b = (const float*)d_in[31];
  const float* lw  = (const float*)d_in[32]; const float* lb  = (const float*)d_in[33];

  const int B = 512;

  // Workspace layout (floats). Big regions ping-pong between conv-pooled
  // activations and projected activations; small MLP buffers live after them.
  float* ws0    = (float*)d_ws;
  float* pooled = ws0;                        // max 512*181*399 = 36,976,128 fl
  float* hbuf   = ws0 + 36974592;             // max 512*181*100 =  9,267,200 fl
  float* sm     = hbuf + 9267200;
  float* e1 = sm;               // 512*700
  float* e2 = e1 + 358400;      // 512*200
  float* e3 = e2 + 102400;      // 512*100
  float* z  = e3 + 51200;       // 512*530
  float* z1 = z  + 271360;      // 512*500
  float* z2 = z1 + 256000;      // 512*100
  float* z3 = z2 + 51200;       // 512*20

  // ---- convolutional tower (fused conv+relu+conv+relu+pool, then WMMA proj)
  dwconv2_relu_pool<<<dim3((399 + 127) / 128, B), 128, 0, stream>>>(
      x, w1, b1, w2, b2, pooled, 365, 399, 181);
  launch_gemm(pooled, p1w, p1b, hbuf, B * 181, 100, 399, 1, stream);   // -> (B,181,100)

  dwconv2_relu_pool<<<dim3(1, B), 128, 0, stream>>>(
      hbuf, w3, b3, w4, b4, pooled, 181, 100, 89);
  launch_gemm(pooled, p2w, p2b, hbuf, B * 89, 50, 100, 1, stream);     // -> (B,89,50)

  dwconv2_relu_pool<<<dim3(1, B), 128, 0, stream>>>(
      hbuf, w5, b5, w6, b6, pooled, 89, 50, 43);
  launch_gemm(pooled, p3w, p3b, hbuf, B * 43, 10, 50, 1, stream);      // enc1 (B,430)

  // ---- x2 MLP branch
  launch_gemm(x2, l1w, l1b, e1, B, 700, 1307, 1, stream);
  launch_gemm(e1, l2w, l2b, e2, B, 200, 700, 1, stream);
  launch_gemm(e2, l3w, l3b, e3, B, 100, 200, 0, stream);               // no relu

  // ---- concat + head
  concat_rows<<<(B * 530 + 255) / 256, 256, 0, stream>>>(hbuf, e3, z, B);
  launch_gemm(z,  c1w, c1b, z1, B, 500, 530, 1, stream);
  launch_gemm(z1, c2w, c2b, z2, B, 100, 500, 1, stream);
  launch_gemm(z2, c3w, c3b, z3, B, 20, 100, 1, stream);
  launch_gemm(z3, lw,  lb, (float*)d_out, B, 1, 20, 0, stream);        // (B,1)
}
